// DistanceEstimator_43834436223740
// MI455X (gfx1250) — matrix operand order repaired and checked by reading, not compiled, not run on hardware
//
#include <hip/hip_runtime.h>
#include <hip/hip_bf16.h>
#include <stdint.h>

typedef __attribute__((ext_vector_type(16))) _Float16 v16h;
typedef __attribute__((ext_vector_type(8)))  float    v8f;

#define RELS 32
#define HD   64
#define ZLD  132   // padded leading dim for regressor input [B, 2H+1]

// ---------------------------------------------------------------------------
// WMMA fragment loaders (CDNA5 16-bit layouts, CDNA5 ISA §7.12.2)
// A (16x32, MxK): lane L -> row M = L&15, kh = L>>4 selects K-halves:
//   elements 0..7  -> K = kh*8 + 0..7
//   elements 8..15 -> K = 16 + kh*8 + 0..7
// B (32x16, KxN) per-lane column data (we keep W transposed [n][k]):
//   lane L -> col N = L&15, same K order as A (kh*8 + i, 16 + kh*8 + i)
// ---------------------------------------------------------------------------
__device__ __forceinline__ v16h load_a_frag(const _Float16* __restrict__ row,
                                            int k0, int kh) {
  union { uint4 q[2]; v16h v; } u;
  const _Float16* p = row + k0 + kh * 8;
  u.q[0] = *(const uint4*)(p);
  u.q[1] = *(const uint4*)(p + 16);
  return u.v;
}
__device__ __forceinline__ v16h load_b_frag(const _Float16* __restrict__ ncol,
                                            int k0, int kh) {
  union { uint4 q[2]; v16h v; } u;
  const _Float16* p = ncol + k0 + kh * 8;
  u.q[0] = *(const uint4*)(p);
  u.q[1] = *(const uint4*)(p + 16);
  return u.v;
}

// ------------------------- conversion kernels ------------------------------
__global__ void cvt_x_f16(const float* __restrict__ x, _Float16* __restrict__ xf,
                          int N, int padN, int K) {
  int idx = blockIdx.x * blockDim.x + threadIdx.x;
  if (idx < padN * K) {
    int i = idx / K;
    xf[idx] = (i < N) ? (_Float16)x[idx] : (_Float16)0.0f;
  }
}

// W [R][K][H] f32 row-major  ->  Wt [R][H][K] f16 (transposed for B frags)
__global__ void cvt_w_f16_t(const float* __restrict__ W, _Float16* __restrict__ Wt,
                            int total, int K, int Hc) {
  int idx = blockIdx.x * blockDim.x + threadIdx.x;
  if (idx < total) {
    int r   = idx / (K * Hc);
    int rem = idx - r * K * Hc;
    int k   = rem / Hc;
    int n   = rem - k * Hc;
    Wt[((size_t)r * Hc + n) * K + k] = (_Float16)W[idx];
  }
}

// ------------------------- edge preprocessing ------------------------------
__global__ void edge_count(const int* __restrict__ dst, const int* __restrict__ type,
                           int E, int* __restrict__ cnt, int* __restrict__ hist) {
  int t = blockIdx.x * blockDim.x + threadIdx.x;
  if (t < E) {
    int r = type[t];
    atomicAdd(&cnt[(size_t)dst[t] * RELS + r], 1);
    atomicAdd(&hist[r], 1);
  }
}

__global__ void scan32(const int* __restrict__ hist, int* __restrict__ off) {
  if (threadIdx.x == 0 && blockIdx.x == 0) {
    int s = 0; off[0] = 0;
    for (int i = 0; i < RELS; ++i) { s += hist[i]; off[i + 1] = s; }
  }
}

__global__ void edge_bucket(const int* __restrict__ src, const int* __restrict__ dst,
                            const int* __restrict__ type, int E,
                            const int* __restrict__ off, int* __restrict__ cursor,
                            int* __restrict__ srcS, int* __restrict__ dstS) {
  int t = blockIdx.x * blockDim.x + threadIdx.x;
  if (t < E) {
    int r = type[t];
    int p = off[r] + atomicAdd(&cursor[r], 1);
    srcS[p] = src[t];
    dstS[p] = dst[t];
  }
}

// ------------------------- dense root GEMM ---------------------------------
// C[padN, 64] = A[padN, K] @ Bt^T + bias  (Bt is [64][K] f16), overwrite.
template <int K>
__global__ __launch_bounds__(256)
void gemm_root(const _Float16* __restrict__ A, const _Float16* __restrict__ Bt,
               const float* __restrict__ bias, float* __restrict__ C) {
  const int lane = threadIdx.x & 31;
  const int wave = threadIdx.x >> 5;
  const int kh   = lane >> 4;
  const int m0   = blockIdx.x * 128 + wave * 16;
  const _Float16* Arow = A + (size_t)(m0 + (lane & 15)) * K;

  v8f acc[4] = {};
#pragma unroll
  for (int k0 = 0; k0 < K; k0 += 32) {
    v16h a = load_a_frag(Arow, k0, kh);
#pragma unroll
    for (int t = 0; t < 4; ++t) {
      v16h b = load_b_frag(Bt + (size_t)(t * 16 + (lane & 15)) * K, k0, kh);
      acc[t] = __builtin_amdgcn_wmma_f32_16x16x32_f16(
          false, a, false, b, (short)0, acc[t], false, false);
    }
  }
  const int mbase = m0 + kh * 8;
#pragma unroll
  for (int t = 0; t < 4; ++t) {
    const int n  = t * 16 + (lane & 15);
    const float bv = bias[n];
#pragma unroll
    for (int v = 0; v < 8; ++v)
      C[(size_t)(mbase + v) * HD + n] = acc[t][v] + bv;
  }
}

// ---------------- gather -> WMMA -> scaled atomic scatter ------------------
// For relation r = blockIdx.y: edges [off[r], off[r+1]) in the bucketed list.
// Each wave: 16 edges per tile. B fragments (this relation's weights) are
// loop-invariant and hoisted into registers; the steady-state loop is
// 8 gathered A loads -> K/32 * 4 WMMAs -> scaled f32 atomic scatter.
template <int K>
__global__ __launch_bounds__(256)
void rel_gemm_scatter(const _Float16* __restrict__ X,
                      const _Float16* __restrict__ Wt,   // [RELS][64][K]
                      const int* __restrict__ off,
                      const int* __restrict__ srcS, const int* __restrict__ dstS,
                      const int* __restrict__ cnt,       // [N][RELS]
                      float* __restrict__ out) {
  constexpr int KS = K / 32;
  const int r      = blockIdx.y;
  const int eStart = off[r];
  const int eEnd   = off[r + 1];
  const int lane   = threadIdx.x & 31;
  const int wave   = threadIdx.x >> 5;
  const int kh     = lane >> 4;
  const _Float16* Btr = Wt + (size_t)r * HD * K;
  const int waveStride = gridDim.x * 8;

  // Hoisted, loop-invariant B fragments (16 v16h for K=128, 8 for K=64).
  v16h bf[4][KS];
#pragma unroll
  for (int t = 0; t < 4; ++t) {
    const _Float16* ncol = Btr + (size_t)(t * 16 + (lane & 15)) * K;
#pragma unroll
    for (int ks = 0; ks < KS; ++ks)
      bf[t][ks] = load_b_frag(ncol, ks * 32, kh);
  }

  for (int tile = blockIdx.x * 8 + wave;; tile += waveStride) {
    const int base = eStart + tile * 16;
    if (base >= eEnd) break;

    const int eA   = base + (lane & 15);
    const int eNext = eA + waveStride * 16;
    if (eNext < eEnd) __builtin_prefetch(&srcS[eNext], 0, 1);

    const int srow = (eA < eEnd) ? srcS[eA] : 0;
    const _Float16* Arow = X + (size_t)srow * K;

    v8f acc[4] = {};
#pragma unroll
    for (int ks = 0; ks < KS; ++ks) {
      v16h a = load_a_frag(Arow, ks * 32, kh);
#pragma unroll
      for (int t = 0; t < 4; ++t)
        acc[t] = __builtin_amdgcn_wmma_f32_16x16x32_f16(
            false, a, false, bf[t][ks], (short)0, acc[t], false, false);
    }
#pragma unroll
    for (int v = 0; v < 8; ++v) {
      const int e = base + kh * 8 + v;
      if (e < eEnd) {
        const int d = dstS[e];
        const int c = cnt[(size_t)d * RELS + r];
        const float w = 1.0f / (float)(c > 1 ? c : 1);
        float* orow = out + (size_t)d * HD;
#pragma unroll
        for (int t = 0; t < 4; ++t)
          atomicAdd(orow + t * 16 + (lane & 15), acc[t][v] * w);
      }
    }
  }
}

// ------------------------- activation / pooling ----------------------------
__global__ void relu_cvt_f16(const float* __restrict__ in, _Float16* __restrict__ o,
                             int n) {
  int i = blockIdx.x * blockDim.x + threadIdx.x;
  if (i < n) { float v = in[i]; o[i] = (_Float16)(v > 0.0f ? v : 0.0f); }
}

__global__ void pool_add(const float* __restrict__ h, const int* __restrict__ batch,
                         float* __restrict__ z, int zoff, int N) {
  int i = blockIdx.x * blockDim.x + threadIdx.x;
  if (i < N * HD) {
    int node = i >> 6, c = i & 63;
    float v = h[i]; v = v > 0.0f ? v : 0.0f;          // fused ReLU of layer 2
    atomicAdd(&z[(size_t)batch[node] * ZLD + zoff + c], v);
  }
}

__global__ void pool_cnt(const int* __restrict__ batch, int* __restrict__ bcnt,
                         int N) {
  int i = blockIdx.x * blockDim.x + threadIdx.x;
  if (i < N) atomicAdd(&bcnt[batch[i]], 1);
}

__global__ void pool_div(float* __restrict__ z, const int* __restrict__ bcnt,
                         int zoff, int B) {
  int idx = blockIdx.x * blockDim.x + threadIdx.x;
  if (idx < B * HD) {
    int b = idx >> 6, c = idx & 63;
    int n = bcnt[b]; if (n < 1) n = 1;
    z[(size_t)b * ZLD + zoff + c] *= (1.0f / (float)n);
  }
}

// ------------------------- depth normalization -----------------------------
__global__ void depth_norm(const float* __restrict__ d, float* __restrict__ z,
                           int B) {
  __shared__ float s1[512], s2[512];
  int t = threadIdx.x;
  float v = (t < B) ? d[t] : 0.0f;
  s1[t] = v; s2[t] = v * v;
  __syncthreads();
  for (int st = 256; st > 0; st >>= 1) {
    if (t < st) { s1[t] += s1[t + st]; s2[t] += s2[t + st]; }
    __syncthreads();
  }
  float mean = s1[0] / (float)B;
  float var  = s2[0] / (float)B - mean * mean;
  float stdv = sqrtf(var > 0.0f ? var : 0.0f);
  if (t < B) z[(size_t)t * ZLD + 2 * HD] = (v - mean) / (stdv + 1e-6f);
}

// ------------------------- regressor head ----------------------------------
__global__ void regressor(const float* __restrict__ z, const float* __restrict__ W1,
                          const float* __restrict__ b1, const float* __restrict__ W2,
                          const float* __restrict__ b2, float* __restrict__ outp) {
  int b = blockIdx.x, h = threadIdx.x;   // 64 threads
  const float* zr = z + (size_t)b * ZLD;
  float acc = b1[h];
  for (int k = 0; k < 2 * HD + 1; ++k) acc += zr[k] * W1[k * HD + h];
  acc = acc > 0.0f ? acc : 0.0f;
  __shared__ float sh[HD];
  sh[h] = acc * W2[h];
  __syncthreads();
  for (int st = 32; st > 0; st >>= 1) {
    if (h < st) sh[h] += sh[h + st];
    __syncthreads();
  }
  if (h == 0) outp[b] = sh[0] + b2[0];
}

// ===========================================================================
extern "C" void kernel_launch(void* const* d_in, const int* in_sizes, int n_in,
                              void* d_out, int out_size, void* d_ws, size_t ws_size,
                              hipStream_t stream) {
  const int N  = in_sizes[3];            // nodes
  const int E  = in_sizes[2];            // edges
  const int IN = in_sizes[0] / N;        // 128
  const int B  = in_sizes[8];            // graphs
  const int padN = ((N + 127) / 128) * 128;

  // ---- workspace carve-up (256B aligned) ----
  char* ws = (char*)d_ws;
  size_t o = 0;
  auto carve = [&](size_t bytes) -> char* {
    char* p = ws + o;
    o += (bytes + 255) & ~(size_t)255;
    return p;
  };
  _Float16* xf16   = (_Float16*)carve((size_t)padN * IN * 2);
  _Float16* h1f16  = (_Float16*)carve((size_t)padN * HD * 2);
  float*    out1   = (float*)   carve((size_t)padN * HD * 4);
  float*    out2   = (float*)   carve((size_t)padN * HD * 4);
  _Float16* wt1    = (_Float16*)carve((size_t)RELS * HD * IN * 2);
  _Float16* wt2    = (_Float16*)carve((size_t)RELS * HD * HD * 2);
  _Float16* rt1    = (_Float16*)carve((size_t)HD * IN * 2);
  _Float16* rt2    = (_Float16*)carve((size_t)HD * HD * 2);
  int*      cnt    = (int*)     carve((size_t)N * RELS * 4);
  int*      hist   = (int*)     carve(RELS * 4);
  int*      off    = (int*)     carve((RELS + 1) * 4);
  int*      cursor = (int*)     carve(RELS * 4);
  int*      srcS   = (int*)     carve((size_t)E * 4);
  int*      dstS   = (int*)     carve((size_t)E * 4);
  int*      bcnt   = (int*)     carve((size_t)B * 4);
  float*    z      = (float*)   carve((size_t)B * ZLD * 4);
  (void)ws_size; (void)n_in;

  const int TB = 256;
  const int gE    = (E + TB - 1) / TB;
  const int gXin  = (padN * IN + TB - 1) / TB;
  const int gNH   = (padN * HD + TB - 1) / TB;
  const int gPool = (N * HD + TB - 1) / TB;

  hipMemsetAsync(z, 0, (size_t)B * ZLD * 4, stream);

  auto run_encoder = [&](const float* x, const int* eidx, const int* etype,
                         const int* batch, const float* W1, const float* Ro1,
                         const float* B1, const float* W2, const float* Ro2,
                         const float* B2, int zoff) {
    const int* src = eidx;
    const int* dst = eidx + E;

    // convert inputs/weights to f16 (weights transposed for B-fragments)
    cvt_x_f16<<<gXin, TB, 0, stream>>>(x, xf16, N, padN, IN);
    cvt_w_f16_t<<<(RELS * IN * HD + TB - 1) / TB, TB, 0, stream>>>(W1, wt1, RELS * IN * HD, IN, HD);
    cvt_w_f16_t<<<(RELS * HD * HD + TB - 1) / TB, TB, 0, stream>>>(W2, wt2, RELS * HD * HD, HD, HD);
    cvt_w_f16_t<<<(IN * HD + TB - 1) / TB, TB, 0, stream>>>(Ro1, rt1, IN * HD, IN, HD);
    cvt_w_f16_t<<<(HD * HD + TB - 1) / TB, TB, 0, stream>>>(Ro2, rt2, HD * HD, HD, HD);

    // bucket edges by relation + per-(dst, r) in-degree for the mean
    hipMemsetAsync(cnt, 0, (size_t)N * RELS * 4, stream);
    hipMemsetAsync(hist, 0, RELS * 4, stream);
    hipMemsetAsync(cursor, 0, RELS * 4, stream);
    edge_count<<<gE, TB, 0, stream>>>(dst, etype, E, cnt, hist);
    scan32<<<1, 32, 0, stream>>>(hist, off);
    edge_bucket<<<gE, TB, 0, stream>>>(src, dst, etype, E, off, cursor, srcS, dstS);

    // ---- layer 1: out1 = x@Root1 + b1  (+ per-relation mean messages) ----
    gemm_root<128><<<padN / 128, TB, 0, stream>>>(xf16, rt1, B1, out1);
    rel_gemm_scatter<128><<<dim3(128, RELS), TB, 0, stream>>>(
        xf16, wt1, off, srcS, dstS, cnt, out1);
    relu_cvt_f16<<<gNH, TB, 0, stream>>>(out1, h1f16, padN * HD);

    // ---- layer 2 ----
    gemm_root<64><<<padN / 128, TB, 0, stream>>>(h1f16, rt2, B2, out2);
    rel_gemm_scatter<64><<<dim3(128, RELS), TB, 0, stream>>>(
        h1f16, wt2, off, srcS, dstS, cnt, out2);

    // ---- global mean pool (ReLU fused into pool_add) ----
    hipMemsetAsync(bcnt, 0, (size_t)B * 4, stream);
    pool_cnt<<<(N + TB - 1) / TB, TB, 0, stream>>>(batch, bcnt, N);
    pool_add<<<gPool, TB, 0, stream>>>(out2, batch, z, zoff, N);
    pool_div<<<(B * HD + TB - 1) / TB, TB, 0, stream>>>(z, bcnt, zoff, B);
  };

  run_encoder((const float*)d_in[0], (const int*)d_in[1], (const int*)d_in[2],
              (const int*)d_in[3], (const float*)d_in[9], (const float*)d_in[10],
              (const float*)d_in[11], (const float*)d_in[12], (const float*)d_in[13],
              (const float*)d_in[14], /*zoff=*/0);

  run_encoder((const float*)d_in[4], (const int*)d_in[5], (const int*)d_in[6],
              (const int*)d_in[7], (const float*)d_in[15], (const float*)d_in[16],
              (const float*)d_in[17], (const float*)d_in[18], (const float*)d_in[19],
              (const float*)d_in[20], /*zoff=*/HD);

  depth_norm<<<1, 512, 0, stream>>>((const float*)d_in[8], z, B);

  regressor<<<B, HD, 0, stream>>>(z, (const float*)d_in[21], (const float*)d_in[22],
                                  (const float*)d_in[23], (const float*)d_in[24],
                                  (float*)d_out);
  (void)out_size;
}
